// SpaceTime_graph_65498251264081
// MI455X (gfx1250) — matrix operand order repaired and checked by reading, not compiled, not run on hardware
//
#include <hip/hip_runtime.h>
#include <hip/hip_bf16.h>

// ---------------------------------------------------------------------------
// B=8, F=12, N=800, D=32, DIM=40, K=20.
// All heavy GEMMs run on v_wmma_f32_16x16x32_f16 with f16 operands that are
// pre-converted once (halves DRAM traffic; staging is pure b128 copies).
// Algebra exploited:
//   * adj2 = M - M^T with M = v1@v2^T  (one GEMM instead of two)
//   * Q / attention / O / final output are batch-independent (a1b is a
//     broadcast) -> computed once, broadcast across the 8-batch output.
//   * a1@a2 turned into A*B^T form by emitting a2^T from the top-k kernel.
// Staging uses GLOBAL_LOAD_ASYNC_TO_LDS_B128 (ASYNCcnt) when available.
// Workspace requirement: ~120 MB.
// ---------------------------------------------------------------------------

#define NB   8
#define NF   12
#define NN   800
#define ND   32
#define NDIM 40
#define KSEL 20
#define NFD  (NF * ND)       // 384
#define NN2  (NN * NN)       // 640000

typedef __attribute__((ext_vector_type(16))) _Float16 v16h;
typedef __attribute__((ext_vector_type(8)))  _Float16 v8h;
typedef __attribute__((ext_vector_type(8)))  float    v8f;
typedef __attribute__((ext_vector_type(4)))  int      v4i;

#if defined(__HIP_DEVICE_COMPILE__) && \
    __has_builtin(__builtin_amdgcn_global_load_async_to_lds_b128) && \
    __has_builtin(__builtin_amdgcn_s_wait_asynccnt)
#define HAVE_ASYNC_LDS 1
typedef __attribute__((address_space(1))) v4i* g_v4i_p;   // global int4*
typedef __attribute__((address_space(3))) v4i* l_v4i_p;   // LDS int4*
#else
#define HAVE_ASYNC_LDS 0
#endif

// ---------------------------------------------------------------------------
// Batched GEMM, all-f16 operands, f32 accumulate:
//   C[b] = epi( A[b] (MxK, row-major f16) x B[b]^T (B is NxK row-major f16) )
// epi: 0 = identity (+bias), 1 = tanh(x*scale), 2 = relu(tanh(scale*x))
// scale = (scale_ptr ? *scale_ptr : scale_const)  (alpha lives on device)
// Optional outputs: Cf (f32) and/or Ch (f16).
// Block: 256 threads = 8 waves; 128x64 tile; each wave owns a 32x32 quadrant
// = 4 WMMA tiles with A/B fragment reuse. K stepped 32 at a time through LDS.
// K must be a multiple of 32 (384 and 800 both are); M/N edges handled by
// clamped (branch-free) loads + guarded stores.
// ---------------------------------------------------------------------------
#define BM 128
#define BN 64
#define BK 32
#define GT 256

__global__ __launch_bounds__(GT)
void gemm_wmma_f16(const _Float16* __restrict__ A, long strideA, int lda,
                   const _Float16* __restrict__ Bm, long strideB, int ldb,
                   float* __restrict__ Cf, _Float16* __restrict__ Ch,
                   long strideC, int ldc,
                   int M, int N, int Kd,
                   int epi, float scale_const, const float* __restrict__ scale_ptr,
                   const float* __restrict__ bias)
{
    __shared__ _Float16 As[BM][BK];   // 8 KB
    __shared__ _Float16 Bs[BN][BK];   // 4 KB

    const int bz = blockIdx.z;
    const _Float16* Ab = A  + (long)bz * strideA;
    const _Float16* Bb = Bm + (long)bz * strideB;

    const int m0 = blockIdx.x * BM;
    const int n0 = blockIdx.y * BN;

    const int tid    = threadIdx.x;
    const int wave   = tid >> 5;
    const int lane   = tid & 31;
    const int wm     = (wave >> 1) << 5;   // 0,32,64,96
    const int wn     = (wave & 1) << 5;    // 0,32
    const int laneM  = lane & 15;
    const int laneHi = lane >> 4;

    v8f acc[2][2] = {};

    for (int k0 = 0; k0 < Kd; k0 += BK) {
        // ---- stage A tile (128x32 halves): 512 b128 chunks, 2 per thread
        #pragma unroll
        for (int c = 0; c < 2; ++c) {
            int ch = tid + c * GT;            // 0..511
            int r  = ch >> 2;                 // 0..127
            int c8 = (ch & 3) << 3;           // 0,8,16,24
            int gm = m0 + r; gm = gm < M ? gm : M - 1;          // clamp, no branch
            v4i* src = (v4i*)(Ab + (long)gm * lda + k0 + c8);
#if HAVE_ASYNC_LDS
            __builtin_amdgcn_global_load_async_to_lds_b128(
                (g_v4i_p)src, (l_v4i_p)(v4i*)&As[r][c8], 0, 0);
#else
            *(v8h*)&As[r][c8] = *(const v8h*)src;
#endif
        }
        // ---- stage B tile (64x32 halves): 256 b128 chunks, 1 per thread
        {
            int r  = tid >> 2;                // 0..63
            int c8 = (tid & 3) << 3;
            int gn = n0 + r; gn = gn < N ? gn : N - 1;
            v4i* src = (v4i*)(Bb + (long)gn * ldb + k0 + c8);
#if HAVE_ASYNC_LDS
            __builtin_amdgcn_global_load_async_to_lds_b128(
                (g_v4i_p)src, (l_v4i_p)(v4i*)&Bs[r][c8], 0, 0);
#else
            *(v8h*)&Bs[r][c8] = *(const v8h*)src;
#endif
        }
#if HAVE_ASYNC_LDS
        __builtin_amdgcn_s_wait_asynccnt(0);
#endif
        __syncthreads();

        // ---- fragments (CDNA5 ISA 7.12.2 wave32 layouts; coalesce to b128)
        v16h af[2], bf[2];
        #pragma unroll
        for (int t = 0; t < 2; ++t) {
            const int mrow = wm + (t << 4) + laneM;
            const int nrow = wn + (t << 4) + laneM;
            #pragma unroll
            for (int j = 0; j < 16; ++j) {
                int ka = (j & 7) + ((j >> 3) << 4) + (laneHi << 3);
                af[t][j] = As[mrow][ka];
                bf[t][j] = Bs[nrow][ka];
            }
        }
        acc[0][0] = __builtin_amdgcn_wmma_f32_16x16x32_f16(false, af[0], false, bf[0], (short)0, acc[0][0], false, false);
        acc[0][1] = __builtin_amdgcn_wmma_f32_16x16x32_f16(false, af[0], false, bf[1], (short)0, acc[0][1], false, false);
        acc[1][0] = __builtin_amdgcn_wmma_f32_16x16x32_f16(false, af[1], false, bf[0], (short)0, acc[1][0], false, false);
        acc[1][1] = __builtin_amdgcn_wmma_f32_16x16x32_f16(false, af[1], false, bf[1], (short)0, acc[1][1], false, false);
        __syncthreads();
    }

    const float scale = scale_ptr ? *scale_ptr : scale_const;
    float* Cfb = Cf ? Cf + (long)bz * strideC : nullptr;
    _Float16* Chb = Ch ? Ch + (long)bz * strideC : nullptr;
    #pragma unroll
    for (int mi = 0; mi < 2; ++mi)
    #pragma unroll
    for (int ni = 0; ni < 2; ++ni)
    #pragma unroll
    for (int r = 0; r < 8; ++r) {
        int gm = m0 + wm + (mi << 4) + r + (laneHi << 3);
        int gn = n0 + wn + (ni << 4) + laneM;
        if (gm < M && gn < N) {
            float x = acc[mi][ni][r];
            if (bias) x += bias[gn];
            if (epi == 1)      x = tanhf(x * scale);
            else if (epi == 2) { x = tanhf(scale * x); x = x > 0.0f ? x : 0.0f; }
            if (Cfb) Cfb[(long)gm * ldc + gn] = x;
            if (Chb) Chb[(long)gm * ldc + gn] = (_Float16)x;
        }
    }
}

// ---------------------------------------------------------------------------
// Pack X (B,F,N,D) f32 -> Y (B, N, F*D) f16 so _sim becomes a GEMM Y*Y^T
__global__ void pack_Y_kernel(const float* __restrict__ X, _Float16* __restrict__ Y)
{
    long i = (long)blockIdx.x * blockDim.x + threadIdx.x;
    long total = (long)NB * NF * NN * ND;
    if (i >= total) return;
    int d = (int)(i % ND); long t = i / ND;
    int n = (int)(t % NN); t /= NN;
    int f = (int)(t % NF); int b = (int)(t / NF);
    Y[((long)b * NN + n) * NFD + f * ND + d] = (_Float16)X[i];
}

__global__ void cvt_f32_f16_kernel(const float* __restrict__ in,
                                   _Float16* __restrict__ out, long n)
{
    long i = (long)blockIdx.x * blockDim.x + threadIdx.x;
    if (i < n) out[i] = (_Float16)in[i];
}

// n_out[n][j] = tanh(alpha * (emb[n,:] . W[j,:] + b[j]))   (800x40, K=40)
__global__ void node_embed_kernel(const float* __restrict__ emb,
                                  const float* __restrict__ W,
                                  const float* __restrict__ bvec,
                                  const float* __restrict__ alphap,
                                  float* __restrict__ out)
{
    int idx = blockIdx.x * blockDim.x + threadIdx.x;
    if (idx >= NN * NDIM) return;
    int n = idx / NDIM, j = idx % NDIM;
    float s = bvec[j];
    #pragma unroll 8
    for (int k = 0; k < NDIM; ++k) s += emb[n * NDIM + k] * W[j * NDIM + k];
    out[idx] = tanhf((*alphap) * s);
}

// adj1 = relu(tanh(alpha * (n1 n2^T - n2 n1^T)))   (800x800, K=40)
__global__ void adj1_kernel(const float* __restrict__ n1, const float* __restrict__ n2,
                            const float* __restrict__ alphap, float* __restrict__ adj1)
{
    int idx = blockIdx.x * blockDim.x + threadIdx.x;
    if (idx >= NN2) return;
    int i = idx / NN, j = idx % NN;
    float s = 0.0f;
    #pragma unroll 8
    for (int k = 0; k < NDIM; ++k)
        s += n1[i * NDIM + k] * n2[j * NDIM + k] - n2[i * NDIM + k] * n1[j * NDIM + k];
    float t = tanhf((*alphap) * s);
    adj1[idx] = t > 0.0f ? t : 0.0f;
}

// Row-wise top-K mask -> f16: out[i,j] = adj[i,j] if (adj+0.01*noise)[i,j]
// is among the K largest in row i, else 0 (threshold = K-th largest).
__global__ void topk_row_kernel(const float* __restrict__ adj,
                                const float* __restrict__ noise,
                                _Float16* __restrict__ out)
{
    int i = blockIdx.x * blockDim.x + threadIdx.x;
    if (i >= NN) return;
    float top[KSEL];
    #pragma unroll
    for (int t = 0; t < KSEL; ++t) top[t] = -3.0e38f;
    for (int j = 0; j < NN; ++j) {
        float s = adj[i * NN + j] + 0.01f * noise[i * NN + j];
        if (s > top[0]) {
            int p = 0;
            while (p < KSEL - 1 && s > top[p + 1]) { top[p] = top[p + 1]; ++p; }
            top[p] = s;
        }
    }
    float thr = top[0];
    for (int j = 0; j < NN; ++j) {
        float s = adj[i * NN + j] + 0.01f * noise[i * NN + j];
        out[i * NN + j] = (s >= thr) ? (_Float16)adj[i * NN + j] : (_Float16)0.0f;
    }
}

// Column-wise top-K over rows (== reference's transposed row top-k), batched.
// Emits masked values as f16 row-major (outRM) and optionally transposed
// (outT) — the transpose converts the later A*B GEMM into A*B^T form.
__global__ void topk_col_kernel(const float* __restrict__ in,
                                const float* __restrict__ noise,
                                _Float16* __restrict__ outRM,
                                _Float16* __restrict__ outT)
{
    int idx = blockIdx.x * blockDim.x + threadIdx.x;
    if (idx >= NB * NN) return;
    int b = idx / NN, j = idx % NN;
    const float* col = in    + (long)b * NN2 + j;
    const float* nz  = noise + (long)b * NN2 + j;
    float top[KSEL];
    #pragma unroll
    for (int t = 0; t < KSEL; ++t) top[t] = -3.0e38f;
    for (int i = 0; i < NN; ++i) {
        float s = col[(long)i * NN] + 0.01f * nz[(long)i * NN];
        if (s > top[0]) {
            int p = 0;
            while (p < KSEL - 1 && s > top[p + 1]) { top[p] = top[p + 1]; ++p; }
            top[p] = s;
        }
    }
    float thr = top[0];
    for (int i = 0; i < NN; ++i) {
        float v = col[(long)i * NN];
        float s = v + 0.01f * nz[(long)i * NN];
        _Float16 hv = (s >= thr) ? (_Float16)v : (_Float16)0.0f;
        outRM[(long)b * NN2 + (long)i * NN + j] = hv;
        if (outT) outT[(long)b * NN2 + (long)j * NN + i] = hv;
    }
}

// adj2 = M - M^T (batched, f32)
__global__ void antisym_kernel(const float* __restrict__ Mb, float* __restrict__ adj2)
{
    long idx = (long)blockIdx.x * blockDim.x + threadIdx.x;
    if (idx >= (long)NB * NN2) return;
    int b = (int)(idx / NN2);
    int r = (int)(idx % NN2);
    int i = r / NN, j = r % NN;
    adj2[idx] = Mb[idx] - Mb[(long)b * NN2 + (long)j * NN + i];
}

// Attention (q-independent): s[k] = Q[n,:].Kp[k,n,:]/sqrt(800); softmax over
// k (8); O[n,e] = sum_k attn[k]*V[k,n,e]. O written as f16 (GEMM input).
__global__ __launch_bounds__(256)
void attn_kernel(const float* __restrict__ Q, const float* __restrict__ Kp,
                 const float* __restrict__ V, _Float16* __restrict__ O)
{
    const int n = blockIdx.x;
    const int tid = threadIdx.x;
    __shared__ float partial[256];
    __shared__ float sk[NB];

    const float* Qn = Q + (long)n * NN;
    for (int k = 0; k < NB; ++k) {
        const float* Kk = Kp + ((long)k * NN + n) * NN;
        float p = 0.0f;
        for (int e = tid; e < NN; e += 256) p += Qn[e] * Kk[e];
        partial[tid] = p;
        __syncthreads();
        for (int s = 128; s > 0; s >>= 1) {
            if (tid < s) partial[tid] += partial[tid + s];
            __syncthreads();
        }
        if (tid == 0) sk[k] = partial[0] * 0.035355339059327376f;  // 1/sqrt(800)
        __syncthreads();
    }
    if (tid == 0) {
        float mx = sk[0];
        for (int k = 1; k < NB; ++k) mx = sk[k] > mx ? sk[k] : mx;
        float sum = 0.0f;
        for (int k = 0; k < NB; ++k) { sk[k] = __expf(sk[k] - mx); sum += sk[k]; }
        float inv = 1.0f / sum;
        for (int k = 0; k < NB; ++k) sk[k] *= inv;
    }
    __syncthreads();
    for (int e = tid; e < NN; e += 256) {
        float o = 0.0f;
        #pragma unroll
        for (int k = 0; k < NB; ++k) o += sk[k] * V[((long)k * NN + n) * NN + e];
        O[(long)n * NN + e] = (_Float16)o;
    }
}

// Broadcast output slab (q=0) to q=1..7 (final result is q-independent).
__global__ void bcast_kernel(float* __restrict__ out)
{
    long i = (long)blockIdx.x * blockDim.x + threadIdx.x;
    if (i >= (long)(NB - 1) * NN2) return;
    out[NN2 + i] = out[i % NN2];
}

// ---------------------------------------------------------------------------
extern "C" void kernel_launch(void* const* d_in, const int* in_sizes, int n_in,
                              void* d_out, int out_size, void* d_ws, size_t ws_size,
                              hipStream_t stream)
{
    const float* Xt     = (const float*)d_in[0];
    const float* Xd     = (const float*)d_in[1];
    const float* alphap = (const float*)d_in[2];
    const float* emb1   = (const float*)d_in[3];
    const float* emb2   = (const float*)d_in[4];
    const float* lin1_w = (const float*)d_in[5];
    const float* lin1_b = (const float*)d_in[6];
    const float* lin2_w = (const float*)d_in[7];
    const float* lin2_b = (const float*)d_in[8];
    const float* Wqkv   = (const float*)d_in[9];
    const float* bqkv   = (const float*)d_in[10];
    const float* Wo     = (const float*)d_in[11];
    const float* bo     = (const float*)d_in[12];
    const float* noise1 = (const float*)d_in[13];
    const float* noise2 = (const float*)d_in[14];
    const float* noise3 = (const float*)d_in[15];
    float* out = (float*)d_out;
    (void)ws_size; (void)in_sizes; (void)n_in; (void)out_size;

    // ---- workspace layout (all offsets 32B-aligned); ~120 MB total
    float* f = (float*)d_ws;
    float* n1    = f;  f += NN * NDIM;            // 32000
    float* n2    = f;  f += NN * NDIM;            // 32000
    float* adj1  = f;  f += NN2;                  // 640000
    float* Qf    = f;  f += NN2;                  // 640000
    float* bufA  = f;  f += (long)NB * NN2;       // M  -> P  -> V
    float* bufB  = f;  f += (long)NB * NN2;       // adj2 -> Kp
    _Float16* h = (_Float16*)f;
    _Float16* Y1h   = h;  h += (long)NB * NN * NFD;
    _Float16* Y2h   = h;  h += (long)NB * NN * NFD;
    _Float16* v1h   = h;  h += (long)NB * NN2;
    _Float16* v2h   = h;  h += (long)NB * NN2;
    _Float16* a1h   = h;  h += NN2;
    _Float16* a2h   = h;  h += (long)NB * NN2;
    _Float16* a2Th  = h;  h += (long)NB * NN2;
    _Float16* a3h   = h;  h += (long)NB * NN2;
    _Float16* Wqkvh = h;  h += 3L * NN2;
    _Float16* Woh   = h;  h += NN2;
    _Float16* Oh    = h;  h += NN2;

    const long packTotal = (long)NB * NF * NN * ND;
    const dim3 gGrid ((NN + BM - 1) / BM, (NN + BN - 1) / BN, NB);
    const dim3 gGrid1((NN + BM - 1) / BM, (NN + BN - 1) / BN, 1);

    // 1. pack features (f32 -> f16) and convert weights once
    pack_Y_kernel<<<(packTotal + 255) / 256, 256, 0, stream>>>(Xt, Y1h);
    pack_Y_kernel<<<(packTotal + 255) / 256, 256, 0, stream>>>(Xd, Y2h);
    cvt_f32_f16_kernel<<<(3L * NN2 + 255) / 256, 256, 0, stream>>>(Wqkv, Wqkvh, 3L * NN2);
    cvt_f32_f16_kernel<<<(NN2 + 255) / 256, 256, 0, stream>>>(Wo, Woh, NN2);

    // 2. node embeddings + adj1 + row top-k -> a1h (f16)
    node_embed_kernel<<<(NN * NDIM + 255) / 256, 256, 0, stream>>>(emb1, lin1_w, lin1_b, alphap, n1);
    node_embed_kernel<<<(NN * NDIM + 255) / 256, 256, 0, stream>>>(emb2, lin2_w, lin2_b, alphap, n2);
    adj1_kernel<<<(NN2 + 255) / 256, 256, 0, stream>>>(n1, n2, alphap, adj1);
    topk_row_kernel<<<(NN + 127) / 128, 128, 0, stream>>>(adj1, noise1, a1h);

    // 3. v1/v2 = tanh(Y Y^T / 12)   [WMMA, batch=8, K=384, f16 out]
    gemm_wmma_f16<<<gGrid, GT, 0, stream>>>(Y1h, (long)NN * NFD, NFD, Y1h, (long)NN * NFD, NFD,
        nullptr, v1h, (long)NN2, NN, NN, NN, NFD, 1, 1.0f / (float)NF, nullptr, nullptr);
    gemm_wmma_f16<<<gGrid, GT, 0, stream>>>(Y2h, (long)NN * NFD, NFD, Y2h, (long)NN * NFD, NFD,
        nullptr, v2h, (long)NN2, NN, NN, NN, NFD, 1, 1.0f / (float)NF, nullptr, nullptr);

    // 4. M = v1 v2^T ; adj2 = M - M^T ; column top-k -> a2h + a2h^T (f16)
    gemm_wmma_f16<<<gGrid, GT, 0, stream>>>(v1h, (long)NN2, NN, v2h, (long)NN2, NN,
        bufA, nullptr, (long)NN2, NN, NN, NN, NN, 0, 1.0f, nullptr, nullptr);
    antisym_kernel<<<(int)(((long)NB * NN2 + 255) / 256), 256, 0, stream>>>(bufA, bufB);
    topk_col_kernel<<<(NB * NN + 127) / 128, 128, 0, stream>>>(bufB, noise2, a2h, a2Th);

    // 5. a3 = relu(tanh(alpha * a1 @ a2)) = a1h * a2Th^T ; column top-k -> a3h
    gemm_wmma_f16<<<gGrid, GT, 0, stream>>>(a1h, 0L, NN, a2Th, (long)NN2, NN,
        bufA, nullptr, (long)NN2, NN, NN, NN, NN, 2, 1.0f, alphap, nullptr);
    topk_col_kernel<<<(NB * NN + 127) / 128, 128, 0, stream>>>(bufA, noise3, a3h, nullptr);

    // 6. projections: Q once; Kp (reuses bufB) and V (reuses bufA) batched
    gemm_wmma_f16<<<gGrid1, GT, 0, stream>>>(a1h, 0L, NN, Wqkvh, 0L, NN,
        Qf, nullptr, 0L, NN, NN, NN, NN, 0, 1.0f, nullptr, bqkv);
    gemm_wmma_f16<<<gGrid, GT, 0, stream>>>(a2h, (long)NN2, NN, Wqkvh + (long)NN2, 0L, NN,
        bufB, nullptr, (long)NN2, NN, NN, NN, NN, 0, 1.0f, nullptr, bqkv + NN);
    gemm_wmma_f16<<<gGrid, GT, 0, stream>>>(a3h, (long)NN2, NN, Wqkvh + 2L * NN2, 0L, NN,
        bufA, nullptr, (long)NN2, NN, NN, NN, NN, 0, 1.0f, nullptr, bqkv + 2 * NN);

    // 7. attention (q-independent) -> Oh (f16)
    attn_kernel<<<NN, 256, 0, stream>>>(Qf, bufB, bufA, Oh);

    // 8. final projection into q=0 slab, then broadcast to q=1..7
    gemm_wmma_f16<<<gGrid1, GT, 0, stream>>>(Oh, 0L, NN, Woh, 0L, NN,
        out, nullptr, 0L, NN, NN, NN, NN, 0, 1.0f, nullptr, bo);
    bcast_kernel<<<(int)(((long)(NB - 1) * NN2 + 255) / 256), 256, 0, stream>>>(out);
}
